// RestrictedParamCountIndividualCayleyLinear_29231547417017
// MI455X (gfx1250) — matrix operand order repaired and checked by reading, not compiled, not run on hardware
//
#include <hip/hip_runtime.h>
#include <hip/hip_bf16.h>
#include <cstdint>
#include <cstddef>

typedef __bf16 bf16_t;
typedef __attribute__((ext_vector_type(16))) __bf16 v16bf;
typedef __attribute__((ext_vector_type(8)))  __bf16 v8bf;
typedef __attribute__((ext_vector_type(8)))  float  v8f;

#define H 512
#define NN 1024
#define APAD 40
#define BPAD 40

#if __has_builtin(__builtin_amdgcn_global_load_async_to_lds_b128) && \
    __has_builtin(__builtin_amdgcn_s_wait_asynccnt)
#define USE_ASYNC 1
#else
#define USE_ASYNC 0
#endif

// Builtin signature (from hipcc diagnostic): param0 = AS1 pointer to
// int __attribute__((vector_size(16))), param1 = AS3 (LDS) pointer.
typedef int v4i __attribute__((__vector_size__(4 * sizeof(int))));
typedef __attribute__((address_space(1))) v4i as1_v4i;
typedef __attribute__((address_space(3))) v4i as3_v4i;

// ---------------------------------------------------------------------------
// Kernel 1: closed-form Woodbury coefficients.
// S = r·1^T - 1·r^T is rank-2  =>  rotation = (I-S)^{-1}(I+S) = I - U·C·V^T
// rotation[i,j] = delta_ij - (a_i + b_i r_j); store ad = d*a, bd = d*b.
// ---------------------------------------------------------------------------
__global__ void k_coeffs(const float* __restrict__ r, const float* __restrict__ d,
                         float* __restrict__ ad, float* __restrict__ bd)
{
    __shared__ float sr[512], sq[512], C[4];
    int t = threadIdx.x;
    float rv = r[t];
    sr[t] = rv; sq[t] = rv * rv;
    __syncthreads();
    for (int off = 256; off > 0; off >>= 1) {
        if (t < off) { sr[t] += sr[t + off]; sq[t] += sq[t + off]; }
        __syncthreads();
    }
    if (t == 0) {
        float s = sr[0], q = sq[0];
        float det = 1.f - s * s + 512.f * q;
        float inv = 1.f / det;
        float g00 = 1.f + s, g11 = 1.f - s;
        C[0] = 1.f + (g00 * g00 - 512.f * q) * inv; // C00
        C[1] = -1024.f * inv;                       // C01
        C[2] = 2.f * q * inv;                       // C10
        C[3] = 1.f + (g11 * g11 - 512.f * q) * inv; // C11
    }
    __syncthreads();
    float a = C[2] - C[0] * rv;
    float b = C[3] - C[1] * rv;
    float dv = d[t];
    ad[t] = dv * a;
    bd[t] = dv * b;
}

// ---------------------------------------------------------------------------
// Kernel 2: lifted[i,j] = sum_o rot90(core, o%4)[i-o, j-o] via closed form.
// Stored row-major as bf16 hi + bf16 lo (hi/lo split of the fp32 value).
// ---------------------------------------------------------------------------
__global__ void k_lifted(const float* __restrict__ r, const float* __restrict__ d,
                         const float* __restrict__ ad, const float* __restrict__ bd,
                         bf16_t* __restrict__ Lh, bf16_t* __restrict__ Ll)
{
    __shared__ float sr[H], sd[H], sa[H], sb[H];
    int t = threadIdx.y * 16 + threadIdx.x;
    sr[t] = r[t];  sr[t + 256] = r[t + 256];
    sd[t] = d[t];  sd[t + 256] = d[t + 256];
    sa[t] = ad[t]; sa[t + 256] = ad[t + 256];
    sb[t] = bd[t]; sb[t + 256] = bd[t + 256];
    __syncthreads();

    int j = blockIdx.x * 16 + threadIdx.x;   // column of lifted (K index of B)
    int i = blockIdx.y * 16 + threadIdx.y;   // row of lifted    (N index of B)
    int olo = max(0, max(i, j) - (H - 1));
    int ohi = min(H, min(i, j));
    float sum = 0.f;
    for (int o = olo; o <= ohi; ++o) {
        int y = i - o, x = j - o, p, q;
        switch (o & 3) {
            case 0:  p = y;         q = x;         break;
            case 1:  p = x;         q = H - 1 - y; break;
            case 2:  p = H - 1 - y; q = H - 1 - x; break;
            default: p = H - 1 - x; q = y;         break;
        }
        float c = -sa[p] - sb[p] * sr[q];
        if (p == q) c += sd[p];
        sum += c;
    }
    bf16_t hi = (bf16_t)sum;
    bf16_t lo = (bf16_t)(sum - (float)hi);
    Lh[(size_t)i * NN + j] = hi;
    Ll[(size_t)i * NN + j] = lo;
}

// ---------------------------------------------------------------------------
// B-tile staging: 128(n) x 32(k) bf16 from lifted[n][k] into padded LDS.
// Async path: GLOBAL_LOAD_ASYNC_TO_LDS_B128 (no VGPR round-trip, ASYNCcnt).
// ---------------------------------------------------------------------------
__device__ __forceinline__ void stage_B(const bf16_t* __restrict__ Lh,
                                        const bf16_t* __restrict__ Ll,
                                        bf16_t* BhBuf, bf16_t* BlBuf,
                                        int t, int n0, int k0)
{
    #pragma unroll
    for (int it = 0; it < 2; ++it) {
        int lin = it * 256 + t;              // 0..511 chunks of 8 bf16
        int n   = lin >> 2;
        int c8  = lin & 3;
        const bf16_t* gh = Lh + (size_t)(n0 + n) * NN + k0 + c8 * 8;
        const bf16_t* gl = Ll + (size_t)(n0 + n) * NN + k0 + c8 * 8;
        bf16_t* lh = BhBuf + n * BPAD + c8 * 8;
        bf16_t* ll = BlBuf + n * BPAD + c8 * 8;
#if USE_ASYNC
        __builtin_amdgcn_global_load_async_to_lds_b128(
            (as1_v4i*)(uintptr_t)gh, (as3_v4i*)(uint32_t)(uintptr_t)lh, 0, 0);
        __builtin_amdgcn_global_load_async_to_lds_b128(
            (as1_v4i*)(uintptr_t)gl, (as3_v4i*)(uint32_t)(uintptr_t)ll, 0, 0);
#else
        *(uint4*)lh = *(const uint4*)gh;
        *(uint4*)ll = *(const uint4*)gl;
#endif
    }
}

// ---------------------------------------------------------------------------
// Kernel 3: out[b,n] = sum_k x[b,k] * lifted[n,k]   (D = A * B, B = lifted^T)
// bf16 hi/lo split: acc += xh*Lh + xh*Ll + xl*Lh  (fp32 accumulate via WMMA)
// 256 threads (8 waves), 128x128 tile, K staged 32/iter; B double-buffered
// and prefetched one iteration ahead via async LDS copies.
// ---------------------------------------------------------------------------
__global__ void __launch_bounds__(256) k_gemm(const float* __restrict__ X,
                                              const bf16_t* __restrict__ Lh,
                                              const bf16_t* __restrict__ Ll,
                                              float* __restrict__ Out)
{
    __shared__ __align__(16) bf16_t Ah[128 * APAD];
    __shared__ __align__(16) bf16_t Al[128 * APAD];
    __shared__ __align__(16) bf16_t Bh[2][128 * BPAD];
    __shared__ __align__(16) bf16_t Bl[2][128 * BPAD];

    const int t    = threadIdx.x;
    const int wave = t >> 5;
    const int lane = t & 31;
    const int half = lane >> 4;   // 0: lanes 0-15, 1: lanes 16-31
    const int l15  = lane & 15;
    const int wn   = wave & 3;    // 4 waves across N (32 cols each)
    const int wm   = wave >> 2;   // 2 waves across M (64 rows each)

    const int m0 = blockIdx.y * 128;
    const int n0 = blockIdx.x * 128;

    v8f acc[4][2];
    #pragma unroll
    for (int a = 0; a < 4; ++a)
        #pragma unroll
        for (int b = 0; b < 2; ++b)
            #pragma unroll
            for (int e = 0; e < 8; ++e) acc[a][b][e] = 0.f;

    // Prologue: kick off B tile for kt=0
    stage_B(Lh, Ll, &Bh[0][0], &Bl[0][0], t, n0, 0);

    for (int kt = 0; kt < 32; ++kt) {
        const int cur = kt & 1;
        const int k0 = kt * 32;

        // Stage A: 128x32 fp32 -> bf16 hi/lo in LDS (on-the-fly split)
        #pragma unroll
        for (int it = 0; it < 4; ++it) {
            int lin = it * 256 + t;              // 0..1023 float4 chunks
            int row = lin >> 3;
            int c4  = lin & 7;
            const float4 v = *(const float4*)(X + (size_t)(m0 + row) * NN + k0 + c4 * 4);
            int base = row * APAD + c4 * 4;
            bf16_t h0 = (bf16_t)v.x, h1 = (bf16_t)v.y, h2 = (bf16_t)v.z, h3 = (bf16_t)v.w;
            Ah[base + 0] = h0; Ah[base + 1] = h1; Ah[base + 2] = h2; Ah[base + 3] = h3;
            Al[base + 0] = (bf16_t)(v.x - (float)h0);
            Al[base + 1] = (bf16_t)(v.y - (float)h1);
            Al[base + 2] = (bf16_t)(v.z - (float)h2);
            Al[base + 3] = (bf16_t)(v.w - (float)h3);
        }

        // Prefetch next B tile into the other buffer, then wait for current.
        // Each wave issues 4 async copies per tile; completion is in-order,
        // so "<= 4 outstanding" == current tile fully landed in LDS.
        if (kt < 31) {
            stage_B(Lh, Ll, &Bh[cur ^ 1][0], &Bl[cur ^ 1][0], t, n0, k0 + 32);
#if USE_ASYNC
            __builtin_amdgcn_s_wait_asynccnt(4);
#endif
        } else {
#if USE_ASYNC
            __builtin_amdgcn_s_wait_asynccnt(0);
#endif
        }
        __syncthreads();

        // B fragments: ISA 16-bit B 32x16 layout: lane col = lane%16,
        // K = {kb..kb+7} U {kb+16..kb+23}, kb = 0 (lanes 0-15) / 8 (lanes 16-31)
        v16bf bhf[2], blf[2];
        #pragma unroll
        for (int ns = 0; ns < 2; ++ns) {
            int col = wn * 32 + ns * 16 + l15;
            const bf16_t* bp = &Bh[cur][col * BPAD + half * 8];
            const bf16_t* lp = &Bl[cur][col * BPAD + half * 8];
            v8bf b0 = *(const v8bf*)bp, b1 = *(const v8bf*)(bp + 16);
            v8bf c0 = *(const v8bf*)lp, c1 = *(const v8bf*)(lp + 16);
            #pragma unroll
            for (int e = 0; e < 8; ++e) {
                bhf[ns][e] = b0[e]; bhf[ns][8 + e] = b1[e];
                blf[ns][e] = c0[e]; blf[ns][8 + e] = c1[e];
            }
        }
        // A fragments: ISA 16-bit A 16x32 layout: lane row = lane%16,
        // K = {kb..kb+7} U {kb+16..kb+23}, kb = 0 (lanes 0-15) / 8 (lanes 16-31)
        #pragma unroll
        for (int ms = 0; ms < 4; ++ms) {
            int row = wm * 64 + ms * 16 + l15;
            const bf16_t* ap = &Ah[row * APAD + half * 8];
            const bf16_t* qp = &Al[row * APAD + half * 8];
            v8bf a0 = *(const v8bf*)ap, a1 = *(const v8bf*)(ap + 16);
            v8bf q0 = *(const v8bf*)qp, q1 = *(const v8bf*)(qp + 16);
            v16bf ah, al;
            #pragma unroll
            for (int e = 0; e < 8; ++e) {
                ah[e] = a0[e]; ah[8 + e] = a1[e];
                al[e] = q0[e]; al[8 + e] = q1[e];
            }
            #pragma unroll
            for (int ns = 0; ns < 2; ++ns) {
                acc[ms][ns] = __builtin_amdgcn_wmma_f32_16x16x32_bf16(
                    false, ah, false, bhf[ns], (short)0, acc[ms][ns], false, false);
                acc[ms][ns] = __builtin_amdgcn_wmma_f32_16x16x32_bf16(
                    false, ah, false, blf[ns], (short)0, acc[ms][ns], false, false);
                acc[ms][ns] = __builtin_amdgcn_wmma_f32_16x16x32_bf16(
                    false, al, false, bhf[ns], (short)0, acc[ms][ns], false, false);
            }
        }
        __syncthreads();
    }

    // Writeback: C/D layout: VGPR v -> row v (lanes 0-15) / v+8 (lanes 16-31)
    #pragma unroll
    for (int ms = 0; ms < 4; ++ms) {
        #pragma unroll
        for (int ns = 0; ns < 2; ++ns) {
            int col   = n0 + wn * 32 + ns * 16 + l15;
            int rbase = m0 + wm * 64 + ms * 16 + half * 8;
            #pragma unroll
            for (int v = 0; v < 8; ++v) {
                Out[(size_t)(rbase + v) * NN + col] = acc[ms][ns][v];
            }
        }
    }
}

// ---------------------------------------------------------------------------
extern "C" void kernel_launch(void* const* d_in, const int* in_sizes, int n_in,
                              void* d_out, int out_size, void* d_ws, size_t ws_size,
                              hipStream_t stream)
{
    const float* x  = (const float*)d_in[0];
    const float* r  = (const float*)d_in[1];
    const float* dg = (const float*)d_in[2];
    float* out = (float*)d_out;

    // workspace layout: ad[512] f32 | bd[512] f32 | Lh[1024^2] bf16 | Ll[1024^2] bf16
    float*  ad = (float*)d_ws;
    float*  bd = ad + H;
    bf16_t* Lh = (bf16_t*)((char*)d_ws + 4096);
    bf16_t* Ll = Lh + (size_t)NN * NN;

    k_coeffs<<<1, 512, 0, stream>>>(r, dg, ad, bd);
    k_lifted<<<dim3(64, 64), dim3(16, 16), 0, stream>>>(r, dg, ad, bd, Lh, Ll);
    k_gemm<<<dim3(8, 512), 256, 0, stream>>>(x, Lh, Ll, out);
}
